// SpatioTemporalAttention_65635690218280
// MI455X (gfx1250) — matrix operand order repaired and checked by reading, not compiled, not run on hardware
//
#include <hip/hip_runtime.h>
#include <hip/hip_bf16.h>
#include <math.h>

// Problem constants (from reference)
#define Bz    32
#define Tz    120
#define Jz    24
#define Ez    128
#define Hz    8
#define Ddim  16
#define Dz    3072       // J*E
#define FFz   256
#define TPAD  128        // T padded to multiple of 16
#define DPAD  32         // head dim padded to WMMA K=32
#define NROWS (Bz*Tz)    // 3840 rows of the (B*T, D) matrices

typedef __attribute__((ext_vector_type(16))) __bf16   v16bf;
typedef __attribute__((ext_vector_type(8)))  float    v8f;
typedef __attribute__((ext_vector_type(4)))  uint32_t u32x4;
typedef __attribute__((ext_vector_type(8)))  uint32_t u32x8;

union AFrag { v16bf v; __bf16 e[16]; };
union CFrag { v8f  v; float  e[8];  };

__device__ __forceinline__ v8f wmma_bf16(v16bf a, v16bf b, v8f c) {
  // D = A(16x32 bf16) * B(32x16 bf16) + C(16x16 f32)
  return __builtin_amdgcn_wmma_f32_16x16x32_bf16(false, a, false, b, (short)0, c, false, false);
}

// ---------------------------------------------------------------------------
// Tensor Data Mover: 2D tile (bf16) from global -> LDS.
// D# per CDNA5 ISA: group0 = {count|flags, lds_addr,
// global_addr lo, global_addr hi | type=2}; group1 carries data_size/dims/strides.
// Issued via inline asm (portable across toolchain builtin arities);
// completion tracked with TENSORcnt.
// ---------------------------------------------------------------------------
__device__ __forceinline__ void tdm_load_2d_bf16(uint32_t lds_byte_addr, const void* gaddr,
                                                 uint32_t tensor_d0, uint32_t tensor_d1,
                                                 uint32_t tile_d0, uint32_t tile_d1,
                                                 uint64_t stride_d0 /* elements */) {
  uint64_t ga = (uint64_t)(uintptr_t)gaddr;
  u32x4 g0;
  g0[0] = 1u;                                   // count=1, user mode
  g0[1] = lds_byte_addr;                        // lds_addr (bytes)
  g0[2] = (uint32_t)ga;                         // global_addr[31:0]
  g0[3] = (uint32_t)(ga >> 32) | (2u << 30);    // global_addr[56:32] | type=2
  u32x8 g1;
  g1[0] = 1u << 16;                             // workgroup_mask=0, data_size=1 (2B)
  g1[1] = (tensor_d0 & 0xFFFFu) << 16;          // abar=0 | tensor_dim0[15:0]
  g1[2] = (tensor_d0 >> 16) | ((tensor_d1 & 0xFFFFu) << 16);
  g1[3] = (tensor_d1 >> 16) | (tile_d0 << 16);  // tensor_dim1 hi | tile_dim0
  g1[4] = tile_d1 & 0xFFFFu;                    // tile_dim1 | tile_dim2=0
  g1[5] = (uint32_t)stride_d0;                  // tensor_dim0_stride[31:0]
  g1[6] = (uint32_t)(stride_d0 >> 32) & 0xFFFFu;// stride hi16 | dim1_stride lo=0
  g1[7] = 0;                                    // dim1_stride hi = 0
  asm volatile("tensor_load_to_lds %0, %1" :: "s"(g0), "s"(g1) : "memory");
}

// ---------------------------------------------------------------------------
// f32 -> bf16 cast / zero
// ---------------------------------------------------------------------------
__global__ void cast_f32_bf16_kernel(const float* __restrict__ in, __bf16* __restrict__ out, int n) {
  int i = blockIdx.x * 256 + threadIdx.x;
  if (i < n) out[i] = (__bf16)in[i];
}

__global__ void zero_f32_kernel(float* __restrict__ p, int n) {
  int i = blockIdx.x * 256 + threadIdx.x;
  if (i < n) p[i] = 0.0f;
}

// ---------------------------------------------------------------------------
// QKV projection: out[t, dd] = sum_e x[b,t,j,e] * W[(h[,j]),dd,e]
// One wave per (b,h,j,t-tile16). M=t (16), N=dd (16), K=E=128 -> 4 WMMA.
// mode 0: temporal layout (b,h,j, TPAD, DPAD) bf16, zero-padded
// mode 1: spatial  layout (b,h,t, J, 16)     bf16
// ---------------------------------------------------------------------------
__global__ void qkv_kernel(const __bf16* __restrict__ x, const __bf16* __restrict__ W,
                           __bf16* __restrict__ out, int per_joint, int mode) {
  int lane = threadIdx.x;
  int idx  = blockIdx.x;
  int ttile = idx & 7;  idx >>= 3;
  int j = idx % Jz;     idx /= Jz;
  int h = idx % Hz;
  int b = idx / Hz;
  int t0 = ttile * 16;
  int n  = lane & 15;
  int hi = (lane >> 4) & 1;

  int tr = t0 + n; if (tr > Tz - 1) tr = Tz - 1;           // clamp OOB pad rows
  const __bf16* pa = x + ((size_t)(b * Tz + tr) * Jz + j) * Ez;

  int wrow = per_joint ? ((h * Jz + j) * Ddim + n) : (h * Ddim + n);
  const __bf16* pw = W + (size_t)wrow * Ez;

  v8f acc = {};
#pragma unroll
  for (int k = 0; k < Ez; k += 32) {
    AFrag fa;
    int ka = k + hi * 8;
#pragma unroll
    for (int i = 0; i < 8; i++) { fa.e[i] = pa[ka + i]; fa.e[8 + i] = pa[ka + 16 + i]; }
    AFrag fb;
    const __bf16* pb = pw + k + hi * 8;
#pragma unroll
    for (int i = 0; i < 8; i++) { fb.e[i] = pb[i]; fb.e[8 + i] = pb[16 + i]; }
    acc = wmma_bf16(fa.v, fb.v, acc);
  }

  CFrag c; c.v = acc;
#pragma unroll
  for (int r = 0; r < 8; r++) {
    int t = t0 + r + hi * 8;
    if (mode == 0) {
      size_t base = (((size_t)(b * Hz + h) * Jz + j) * TPAD + t) * DPAD;
      out[base + n]      = (t < Tz) ? (__bf16)c.e[r] : (__bf16)0.0f;
      out[base + 16 + n] = (__bf16)0.0f;                  // pad d 16..31
    } else {
      if (t < Tz) {
        size_t base = (((size_t)(b * Hz + h) * Tz + t) * Jz + j) * Ddim;
        out[base + n] = (__bf16)c.e[r];
      }
    }
  }
}

// ---------------------------------------------------------------------------
// Temporal attention: per (b,h,j, t-tile16). V tile (128x32) staged into LDS
// via TDM at kernel entry (overlaps score WMMAs + softmax). Scores 16x128 via
// 8 WMMA (K=DPAD), causal softmax in registers, P->LDS (C-layout -> A-layout),
// then s_wait_tensorcnt drains the V stage before the P*V WMMA loop.
// LDS map (static LDS starts at offset 0): [P (16x128) | Vstage (128x32)]
// ---------------------------------------------------------------------------
__global__ void temporal_attn_kernel(const __bf16* __restrict__ qt, const __bf16* __restrict__ kt,
                                     const __bf16* __restrict__ vt, __bf16* __restrict__ ta_pre) {
  __shared__ __bf16 sm[16 * TPAD + TPAD * DPAD];
  __bf16* P  = sm;                    // 16x128 probabilities
  __bf16* Vs = sm + 16 * TPAD;        // 128x32 staged V tile

  int lane = threadIdx.x;
  int idx  = blockIdx.x;
  int ttile = idx & 7;  idx >>= 3;
  int j = idx % Jz;     idx /= Jz;
  int h = idx % Hz;
  int b = idx / Hz;
  int t0 = ttile * 16;
  int n  = lane & 15;
  int hi = (lane >> 4) & 1;

  size_t base = ((size_t)(b * Hz + h) * Jz + j) * TPAD * DPAD;
  const __bf16* Q = qt + base;
  const __bf16* K = kt + base;
  const __bf16* V = vt + base;

  // kick off async V stage: linear 128x32 bf16 tile -> LDS (8KB)
  tdm_load_2d_bf16((uint32_t)(16 * TPAD * 2), V,
                   (uint32_t)(TPAD * DPAD), 1u, (uint32_t)(TPAD * DPAD), 1u,
                   (uint64_t)(TPAD * DPAD));

  AFrag fq;
  {
    const __bf16* p = Q + (size_t)(t0 + n) * DPAD;
    int k0 = hi * 8;
#pragma unroll
    for (int i = 0; i < 8; i++) { fq.e[i] = p[k0 + i]; fq.e[8 + i] = p[k0 + 16 + i]; }
  }

  CFrag s[8];
#pragma unroll
  for (int st = 0; st < 8; st++) {
    AFrag fb;  // B[kk, ss] = K[s0+ss, kk] -> contiguous per-lane row read
    const __bf16* p = K + (size_t)(st * 16 + n) * DPAD + hi * 8;
#pragma unroll
    for (int i = 0; i < 8; i++) { fb.e[i] = p[i]; fb.e[8 + i] = p[16 + i]; }
    v8f z = {};
    s[st].v = wmma_bf16(fq.v, fb.v, z);
  }

  const float sc = 0.25f; // 1/sqrt(16)
#pragma unroll
  for (int r = 0; r < 8; r++) {
    int t = t0 + r + hi * 8;
    float m = -1e30f;
#pragma unroll
    for (int st = 0; st < 8; st++) {
      int ss = st * 16 + n;
      float val = (ss <= t && ss < Tz) ? s[st].e[r] * sc : -1e30f;
      s[st].e[r] = val;
      m = fmaxf(m, val);
    }
#pragma unroll
    for (int o = 1; o < 16; o <<= 1) m = fmaxf(m, __shfl_xor(m, o, 32));
    float sum = 0.0f;
#pragma unroll
    for (int st = 0; st < 8; st++) {
      float p = __expf(s[st].e[r] - m);
      s[st].e[r] = p;
      sum += p;
    }
#pragma unroll
    for (int o = 1; o < 16; o <<= 1) sum += __shfl_xor(sum, o, 32);
    float inv = 1.0f / sum;
#pragma unroll
    for (int st = 0; st < 8; st++)
      P[(r + hi * 8) * TPAD + st * 16 + n] = (__bf16)(s[st].e[r] * inv);
  }
  __syncthreads();
  __builtin_amdgcn_s_wait_tensorcnt(0);   // V tile resident in LDS

  v8f acc = {};
#pragma unroll
  for (int k = 0; k < TPAD; k += 32) {
    AFrag fa;
    const __bf16* p = P + (size_t)n * TPAD;
    int k0 = k + hi * 8;
#pragma unroll
    for (int i = 0; i < 8; i++) { fa.e[i] = p[k0 + i]; fa.e[8 + i] = p[k0 + 16 + i]; }
    AFrag fb;  // B[kk, dd] = Vs[kk, dd] -> strided per-lane column read from LDS
#pragma unroll
    for (int i = 0; i < 8; i++) { fb.e[i] = Vs[(size_t)(k + hi * 8 + i) * DPAD + n]; fb.e[8 + i] = Vs[(size_t)(k + 16 + hi * 8 + i) * DPAD + n]; }
    acc = wmma_bf16(fa.v, fb.v, acc);
  }

  CFrag c; c.v = acc;
#pragma unroll
  for (int r = 0; r < 8; r++) {
    int t = t0 + r + hi * 8;
    if (t < Tz)
      ta_pre[((size_t)(b * Tz) + t) * Dz + h * (Jz * Ddim) + j * Ddim + n] = (__bf16)c.e[r];
  }
}

// ---------------------------------------------------------------------------
// Spatial attention: tiny 24x24 per (b,h,t); f32 VALU with LDS-staged K/V.
// ---------------------------------------------------------------------------
__global__ void spatial_attn_kernel(const __bf16* __restrict__ qs, const __bf16* __restrict__ ks,
                                    const __bf16* __restrict__ vs, __bf16* __restrict__ sa_pre) {
  __shared__ float Ksh[Jz * Ddim];
  __shared__ float Vsh[Jz * Ddim];
  int lane = threadIdx.x;
  int idx  = blockIdx.x;
  int t = idx % Tz; idx /= Tz;
  int h = idx % Hz;
  int b = idx / Hz;
  size_t base = ((size_t)(b * Hz + h) * Tz + t) * Jz * Ddim;

  for (int i = lane; i < Jz * Ddim; i += 32) {
    Ksh[i] = (float)ks[base + i];
    Vsh[i] = (float)vs[base + i];
  }
  __syncthreads();

  if (lane < Jz) {
    float q[Ddim];
#pragma unroll
    for (int dd = 0; dd < Ddim; dd++) q[dd] = (float)qs[base + lane * Ddim + dd];
    float sarr[Jz];
    float m = -1e30f;
    for (int k = 0; k < Jz; k++) {
      float s = 0.0f;
#pragma unroll
      for (int dd = 0; dd < Ddim; dd++) s += q[dd] * Ksh[k * Ddim + dd];
      s *= 0.25f;
      sarr[k] = s;
      m = fmaxf(m, s);
    }
    float sum = 0.0f;
    for (int k = 0; k < Jz; k++) { float p = __expf(sarr[k] - m); sarr[k] = p; sum += p; }
    float inv = 1.0f / sum;
    size_t obase = ((size_t)(b * Tz) + t) * Dz + h * (Jz * Ddim) + lane * Ddim;
#pragma unroll
    for (int dd = 0; dd < Ddim; dd++) {
      float o = 0.0f;
      for (int k = 0; k < Jz; k++) o += sarr[k] * Vsh[k * Ddim + dd];
      sa_pre[obase + dd] = (__bf16)(o * inv);
    }
  }
}

// ---------------------------------------------------------------------------
// TDM-staged GEMM: out[n, co] = sum_k A[n,k] * W[co,k]  (+bias +residual, ReLU)
// One wave per 32(M) x 64(N) tile. Per K-stage (64): two tensor_load_to_lds
// DMAs (A strip 32x64, W tile 64x64) into double-buffered LDS, issued one
// stage ahead; s_wait_tensorcnt drains the previous stage; 16 WMMA consume
// fragments from LDS via ds_load.
// LDS map (static LDS starts at offset 0): [A0 | A1 | W0 | W1]
// ---------------------------------------------------------------------------
#define GA_ELEMS (32 * 64)
#define GW_ELEMS (64 * 64)

__global__ void gemm_tdm_kernel(const __bf16* __restrict__ A, const __bf16* __restrict__ W,
                                const float* __restrict__ bias, const float* __restrict__ res,
                                float* __restrict__ outF, __bf16* __restrict__ outB,
                                int Ncols, int Kdim, int relu) {
  __shared__ __bf16 smem[2 * GA_ELEMS + 2 * GW_ELEMS];

  int lane = threadIdx.x;
  int c0 = blockIdx.x * 64;
  int m0 = blockIdx.y * 32;
  int n  = lane & 15;
  int hi = (lane >> 4) & 1;

  const __bf16* Abase = A + (size_t)m0 * Kdim;
  const __bf16* Wbase = W + (size_t)c0 * Kdim;

  v8f acc[2][4];
#pragma unroll
  for (int mt = 0; mt < 2; mt++)
#pragma unroll
    for (int sub = 0; sub < 4; sub++) acc[mt][sub] = (v8f){};

  const int nk = Kdim / 64;
  // prologue: stage 0 into buffer 0
  tdm_load_2d_bf16(0u, Abase, (uint32_t)Kdim, 32u, 64u, 32u, (uint64_t)Kdim);
  tdm_load_2d_bf16((uint32_t)(4 * GA_ELEMS), Wbase, (uint32_t)Kdim, 64u, 64u, 64u, (uint64_t)Kdim);

  for (int ks = 0; ks < nk; ks++) {
    int buf = ks & 1;
    if (ks + 1 < nk) {
      uint32_t nb = (uint32_t)((ks + 1) & 1);
      int kn = (ks + 1) * 64;
      tdm_load_2d_bf16(nb * (uint32_t)(2 * GA_ELEMS), Abase + kn,
                       (uint32_t)Kdim, 32u, 64u, 32u, (uint64_t)Kdim);
      tdm_load_2d_bf16((uint32_t)(4 * GA_ELEMS) + nb * (uint32_t)(2 * GW_ELEMS), Wbase + kn,
                       (uint32_t)Kdim, 64u, 64u, 64u, (uint64_t)Kdim);
      __builtin_amdgcn_s_wait_tensorcnt(2);   // stage ks complete; ks+1 in flight
    } else {
      __builtin_amdgcn_s_wait_tensorcnt(0);   // final stage complete
    }

    const __bf16* tA = smem + buf * GA_ELEMS;
    const __bf16* tW = smem + 2 * GA_ELEMS + buf * GW_ELEMS;
#pragma unroll
    for (int ksub = 0; ksub < 64; ksub += 32) {
      AFrag fa[2];
#pragma unroll
      for (int mt = 0; mt < 2; mt++) {
        const __bf16* p = tA + (size_t)(mt * 16 + n) * 64;
        int k0 = ksub + hi * 8;
#pragma unroll
        for (int i = 0; i < 8; i++) { fa[mt].e[i] = p[k0 + i]; fa[mt].e[8 + i] = p[k0 + 16 + i]; }
      }
#pragma unroll
      for (int sub = 0; sub < 4; sub++) {
        AFrag fb;
        const __bf16* p = tW + (size_t)(sub * 16 + n) * 64 + ksub + hi * 8;
#pragma unroll
        for (int i = 0; i < 8; i++) { fb.e[i] = p[i]; fb.e[8 + i] = p[16 + i]; }
        acc[0][sub] = wmma_bf16(fa[0].v, fb.v, acc[0][sub]);
        acc[1][sub] = wmma_bf16(fa[1].v, fb.v, acc[1][sub]);
      }
    }
  }

#pragma unroll
  for (int mt = 0; mt < 2; mt++) {
    CFrag c[4];
#pragma unroll
    for (int sub = 0; sub < 4; sub++) c[sub].v = acc[mt][sub];
#pragma unroll
    for (int sub = 0; sub < 4; sub++) {
#pragma unroll
      for (int r = 0; r < 8; r++) {
        int row = m0 + mt * 16 + r + hi * 8;
        int col = c0 + sub * 16 + n;
        float v = c[sub].e[r] + (bias ? bias[col] : 0.0f);
        if (res) v += res[(size_t)row * Ncols + col];
        if (relu) v = fmaxf(v, 0.0f);
        size_t o = (size_t)row * Ncols + col;
        if (outF) outF[o] = v;
        if (outB) outB[o] = (__bf16)v;
      }
    }
  }
}

// ---------------------------------------------------------------------------
// BatchNorm partial sums + fused applies
// ---------------------------------------------------------------------------
__global__ void bn_partial_kernel(const float* __restrict__ x, float* __restrict__ sums) {
  int c  = blockIdx.x * 256 + threadIdx.x;   // channel
  int r0 = blockIdx.y * 128;
  float s = 0.0f, s2 = 0.0f;
  for (int nn = r0; nn < r0 + 128; nn++) {
    float v = x[(size_t)nn * Dz + c];
    s += v; s2 += v * v;
  }
  atomicAdd(&sums[c], s);
  atomicAdd(&sums[Dz + c], s2);
}

__global__ void att_fuse_kernel(const float* __restrict__ pt, const float* __restrict__ ps,
                                const float* __restrict__ sum_t, const float* __restrict__ sum_s,
                                const float* __restrict__ g_t, const float* __restrict__ b_t,
                                const float* __restrict__ g_s, const float* __restrict__ b_s,
                                float* __restrict__ attF, __bf16* __restrict__ attB) {
  size_t i = (size_t)blockIdx.x * 256 + threadIdx.x;
  if (i >= (size_t)NROWS * Dz) return;
  int c = (int)(i % Dz);
  const float invN = 1.0f / (float)NROWS;
  float mt = sum_t[c] * invN;
  float vt = sum_t[Dz + c] * invN - mt * mt;
  float ms = sum_s[c] * invN;
  float vs = sum_s[Dz + c] * invN - ms * ms;
  float a = (pt[i] - mt) * rsqrtf(vt + 1e-5f) * g_t[c] + b_t[c];
  float s = (ps[i] - ms) * rsqrtf(vs + 1e-5f) * g_s[c] + b_s[c];
  float o = a + s;
  attF[i] = o;
  attB[i] = (__bf16)o;
}

__global__ void bn_apply_kernel(const float* __restrict__ x, const float* __restrict__ sums,
                                const float* __restrict__ g, const float* __restrict__ bb,
                                float* __restrict__ out) {
  size_t i = (size_t)blockIdx.x * 256 + threadIdx.x;
  if (i >= (size_t)NROWS * Dz) return;
  int c = (int)(i % Dz);
  const float invN = 1.0f / (float)NROWS;
  float m = sums[c] * invN;
  float v = sums[Dz + c] * invN - m * m;
  out[i] = (x[i] - m) * rsqrtf(v + 1e-5f) * g[c] + bb[c];
}

// ---------------------------------------------------------------------------
// Host launch
// ---------------------------------------------------------------------------
extern "C" void kernel_launch(void* const* d_in, const int* in_sizes, int n_in,
                              void* d_out, int out_size, void* d_ws, size_t ws_size,
                              hipStream_t stream) {
  const float* src  = (const float*)d_in[0];
  const float* Wq_t = (const float*)d_in[1];
  const float* Wk_t = (const float*)d_in[2];
  const float* Wv_t = (const float*)d_in[3];
  const float* Wp_t = (const float*)d_in[4];
  const float* bp_t = (const float*)d_in[5];
  const float* g_t  = (const float*)d_in[6];
  const float* b_t  = (const float*)d_in[7];
  const float* Wq_s = (const float*)d_in[8];
  const float* Wk_s = (const float*)d_in[9];
  const float* Wv_s = (const float*)d_in[10];
  const float* Wp_s = (const float*)d_in[11];
  const float* bp_s = (const float*)d_in[12];
  const float* g_s  = (const float*)d_in[13];
  const float* b_s  = (const float*)d_in[14];
  const float* W1   = (const float*)d_in[15];
  const float* b1   = (const float*)d_in[16];
  const float* W2   = (const float*)d_in[17];
  const float* b2   = (const float*)d_in[18];
  const float* g_f  = (const float*)d_in[19];
  const float* b_f  = (const float*)d_in[20];
  float* out = (float*)d_out;

  // workspace carve-out
  char* w = (char*)d_ws;
  auto alloc = [&](size_t bytes) -> char* {
    char* p = w;
    w += (bytes + 255) & ~(size_t)255;
    return p;
  };
  const size_t NX   = (size_t)Bz * Tz * Jz * Ez;
  const size_t NWPJ = (size_t)Hz * Jz * Ddim * Ez;
  const size_t NWSH = (size_t)Hz * Ddim * Ez;
  const size_t NWP  = (size_t)Dz * Dz;
  const size_t NW1  = (size_t)FFz * Dz;
  const size_t NQT  = (size_t)Bz * Hz * Jz * TPAD * DPAD;
  const size_t NQS  = (size_t)Bz * Hz * Tz * Jz * Ddim;
  const size_t NMAT = (size_t)NROWS * Dz;
  const size_t NH1  = (size_t)NROWS * FFz;

  __bf16* xbf  = (__bf16*)alloc(NX * 2);
  __bf16* wqt  = (__bf16*)alloc(NWPJ * 2);
  __bf16* wkt  = (__bf16*)alloc(NWPJ * 2);
  __bf16* wvt  = (__bf16*)alloc(NWPJ * 2);
  __bf16* wqs  = (__bf16*)alloc(NWPJ * 2);
  __bf16* wks  = (__bf16*)alloc(NWSH * 2);
  __bf16* wvs  = (__bf16*)alloc(NWSH * 2);
  __bf16* wpt  = (__bf16*)alloc(NWP * 2);
  __bf16* wps  = (__bf16*)alloc(NWP * 2);
  __bf16* w1b  = (__bf16*)alloc(NW1 * 2);
  __bf16* w2b  = (__bf16*)alloc(NW1 * 2);
  __bf16* qt   = (__bf16*)alloc(NQT * 2);
  __bf16* kt   = (__bf16*)alloc(NQT * 2);
  __bf16* vt   = (__bf16*)alloc(NQT * 2);
  __bf16* qs   = (__bf16*)alloc(NQS * 2);
  __bf16* ks   = (__bf16*)alloc(NQS * 2);
  __bf16* vs   = (__bf16*)alloc(NQS * 2);
  __bf16* tap  = (__bf16*)alloc(NMAT * 2);
  __bf16* sap  = (__bf16*)alloc(NMAT * 2);
  float*  pt   = (float*)alloc(NMAT * 4);
  float*  ps   = (float*)alloc(NMAT * 4);
  float*  attF = (float*)alloc(NMAT * 4);
  __bf16* attB = (__bf16*)alloc(NMAT * 2);
  __bf16* h1b  = (__bf16*)alloc(NH1 * 2);
  float*  ffp  = (float*)alloc(NMAT * 4);
  float*  sums = (float*)alloc((size_t)6 * Dz * 4);
  float* sum_t = sums;
  float* sum_s = sums + 2 * Dz;
  float* sum_f = sums + 4 * Dz;

  auto cast = [&](const float* in, __bf16* o, size_t n) {
    cast_f32_bf16_kernel<<<dim3((unsigned)((n + 255) / 256)), dim3(256), 0, stream>>>(in, o, (int)n);
  };
  cast(src, xbf, NX);
  cast(Wq_t, wqt, NWPJ);  cast(Wk_t, wkt, NWPJ);  cast(Wv_t, wvt, NWPJ);
  cast(Wq_s, wqs, NWPJ);  cast(Wk_s, wks, NWSH);  cast(Wv_s, wvs, NWSH);
  cast(Wp_t, wpt, NWP);   cast(Wp_s, wps, NWP);
  cast(W1, w1b, NW1);     cast(W2, w2b, NW1);

  // QKV projections
  dim3 qg((unsigned)(Bz * Hz * Jz * (TPAD / 16)));
  qkv_kernel<<<qg, dim3(32), 0, stream>>>(xbf, wqt, qt, 1, 0);
  qkv_kernel<<<qg, dim3(32), 0, stream>>>(xbf, wkt, kt, 1, 0);
  qkv_kernel<<<qg, dim3(32), 0, stream>>>(xbf, wvt, vt, 1, 0);
  qkv_kernel<<<qg, dim3(32), 0, stream>>>(xbf, wqs, qs, 1, 1);
  qkv_kernel<<<qg, dim3(32), 0, stream>>>(xbf, wks, ks, 0, 1);
  qkv_kernel<<<qg, dim3(32), 0, stream>>>(xbf, wvs, vs, 0, 1);

  // attention cores
  temporal_attn_kernel<<<qg, dim3(32), 0, stream>>>(qt, kt, vt, tap);
  spatial_attn_kernel<<<dim3((unsigned)(Bz * Hz * Tz)), dim3(32), 0, stream>>>(qs, ks, vs, sap);

  // output projections (+bias +residual), TDM-staged
  dim3 gP(Dz / 64, NROWS / 32);
  gemm_tdm_kernel<<<gP, dim3(32), 0, stream>>>(tap, wpt, bp_t, src, pt, (__bf16*)nullptr, Dz, Dz, 0);
  gemm_tdm_kernel<<<gP, dim3(32), 0, stream>>>(sap, wps, bp_s, src, ps, (__bf16*)nullptr, Dz, Dz, 0);

  // BN stats for temporal/spatial branches
  zero_f32_kernel<<<dim3((6 * Dz + 255) / 256), dim3(256), 0, stream>>>(sums, 6 * Dz);
  dim3 bnG(Dz / 256, NROWS / 128);
  bn_partial_kernel<<<bnG, dim3(256), 0, stream>>>(pt, sum_t);
  bn_partial_kernel<<<bnG, dim3(256), 0, stream>>>(ps, sum_s);

  // att = BN(pt) + BN(ps)
  unsigned nel = (unsigned)((NMAT + 255) / 256);
  att_fuse_kernel<<<dim3(nel), dim3(256), 0, stream>>>(pt, ps, sum_t, sum_s,
                                                       g_t, b_t, g_s, b_s, attF, attB);

  // FF: h1 = relu(att @ W1^T + b1); ffp = h1 @ W2^T + b2 + att
  dim3 gF1(FFz / 64, NROWS / 32);
  gemm_tdm_kernel<<<gF1, dim3(32), 0, stream>>>(attB, w1b, b1, (const float*)nullptr,
                                                (float*)nullptr, h1b, FFz, Dz, 1);
  gemm_tdm_kernel<<<gP, dim3(32), 0, stream>>>(h1b, w2b, b2, attF, ffp, (__bf16*)nullptr, Dz, FFz, 0);

  // final BN
  bn_partial_kernel<<<bnG, dim3(256), 0, stream>>>(ffp, sum_f);
  bn_apply_kernel<<<dim3(nel), dim3(256), 0, stream>>>(ffp, sum_f, g_f, b_f, out);
}
